// TNPKR_16028817949094
// MI455X (gfx1250) — compile-verified, matrix operand order checked by reading, not run on hardware
//
#include <hip/hip_runtime.h>
#include <hip/hip_bf16.h>
#include <math.h>

// ---------------- problem constants ----------------
#define Lb   6
#define Hh   8
#define Dd   128
#define DHd  16
#define DFFd 512
#define BHh  16
#define Bb   4
#define NQn  512
#define NKn  512
#define NROW 2048   // B*NQ (= B*NK)
#define NALL 4096   // test rows + ctx rows

typedef __attribute__((ext_vector_type(16))) _Float16 v16h;
typedef __attribute__((ext_vector_type(8)))  _Float16 v8h;
typedef __attribute__((ext_vector_type(4)))  _Float16 v4h;
typedef __attribute__((ext_vector_type(8)))  float    v8f;

__device__ __forceinline__ float gelu_f(float x) {
  const float c0 = 0.7978845608028654f;              // sqrt(2/pi)
  float t = tanhf(c0 * (x + 0.044715f * x * x * x)); // jax.nn.gelu (tanh approx)
  return 0.5f * x * (1.0f + t);
}

// ---------------- generic batched WMMA GEMM ----------------
// C = act(alpha * A@B [+bias] [+=C if ACCC] [+resid]) with optional key-mask->-1e9
// A row-major [M,K]; B row-major [K,N] or (TRANSB) [N,K]; C row-major [M,N].
// grid.z = nB*nH batches, per-batch strides (sAb,sAh) etc.
// Staging is float4-vectorized: global_load_b128 -> cvt f16 -> ds_store_b64.
#define TILE_M 64
#define TILE_N 64
#define TILE_K 32
#define LDSR   40   // half-element row stride: 80B (multiple of 16B)

template<int ACT, bool TRANSB, bool ACCC>
__global__ __launch_bounds__(128)
void gemm_wmma(const float* __restrict__ A, const float* __restrict__ Bm,
               const float* __restrict__ bias, const float* __restrict__ resid,
               const unsigned char* __restrict__ mask, int maskStride,
               float* __restrict__ C,
               int M, int N, int K, int lda, int ldb, int ldc,
               long sAb, long sAh, long sBb, long sBh, long sCb, long sCh,
               int nH, float alpha)
{
  __shared__ _Float16 As[TILE_M * LDSR];
  __shared__ _Float16 Bs[TILE_N * LDSR];

  const int tid  = threadIdx.x;
  const int lane = tid & 31;
  const int wave = tid >> 5;
  const int wr   = wave >> 1;   // wave row (0..1) -> 32 rows each
  const int wc   = wave & 1;    // wave col (0..1) -> 32 cols each
  const int g    = lane >> 4;   // half-wave group
  const int r    = lane & 15;

  const int z  = blockIdx.z;
  const int bb = z / nH;
  const int hh = z - bb * nH;
  const long Aoff = (long)bb * sAb + (long)hh * sAh;
  const long Boff = (long)bb * sBb + (long)hh * sBh;
  const long Coff = (long)bb * sCb + (long)hh * sCh;

  const int tileM = blockIdx.y * TILE_M;
  const int tileN = blockIdx.x * TILE_N;
  const bool fullM = (tileM + TILE_M <= M);
  const bool fullN = (tileN + TILE_N <= N);

  v8f acc[2][2];
  v8f zero8;
#pragma unroll
  for (int e = 0; e < 8; e++) zero8[e] = 0.0f;
  acc[0][0] = zero8; acc[0][1] = zero8; acc[1][0] = zero8; acc[1][1] = zero8;

  for (int k0 = 0; k0 < K; k0 += TILE_K) {
    // ---- stage A tile [64 x 32]: float4 loads, ds_store_b64 ----
    // 64 rows * 8 vec4 = 512 slots, 128 threads -> 4 iterations
#pragma unroll
    for (int i = tid; i < TILE_M * (TILE_K / 4); i += 128) {
      int m  = i >> 3;
      int kc = (i & 7) << 2;
      int gm = tileM + m, gk = k0 + kc;
      float4 v = make_float4(0.f, 0.f, 0.f, 0.f);
      if (fullM || gm < M) {
        const float* src = &A[Aoff + (long)gm * lda + gk];
        if (gk + 3 < K) {
          v = *(const float4*)src;                 // global_load_b128 (16B aligned)
        } else if (gk < K) {                       // K is a multiple of 4 in practice
          v.x = src[0];
          if (gk + 1 < K) v.y = src[1];
          if (gk + 2 < K) v.z = src[2];
        }
      }
      v4h h4;
      h4[0] = (_Float16)v.x; h4[1] = (_Float16)v.y;
      h4[2] = (_Float16)v.z; h4[3] = (_Float16)v.w;
      *(v4h*)&As[m * LDSR + kc] = h4;              // ds_store_b64 (8B aligned)
    }

    // ---- stage B tile as [n][k] (K-contiguous in LDS) ----
    if (TRANSB) {
      // B is [N,K] row-major: contiguous along k -> same pattern as A
#pragma unroll
      for (int i = tid; i < TILE_N * (TILE_K / 4); i += 128) {
        int n  = i >> 3;
        int kc = (i & 7) << 2;
        int gn = tileN + n, gk = k0 + kc;
        float4 v = make_float4(0.f, 0.f, 0.f, 0.f);
        if (fullN || gn < N) {
          const float* src = &Bm[Boff + (long)gn * ldb + gk];
          if (gk + 3 < K) {
            v = *(const float4*)src;
          } else if (gk < K) {
            v.x = src[0];
            if (gk + 1 < K) v.y = src[1];
            if (gk + 2 < K) v.z = src[2];
          }
        }
        v4h h4;
        h4[0] = (_Float16)v.x; h4[1] = (_Float16)v.y;
        h4[2] = (_Float16)v.z; h4[3] = (_Float16)v.w;
        *(v4h*)&Bs[n * LDSR + kc] = h4;
      }
    } else {
      // B is [K,N] row-major: contiguous along n -> float4 along n, scatter 4 rows
      // 32 k * 16 vec4 = 512 slots, 128 threads -> 4 iterations
#pragma unroll
      for (int i = tid; i < TILE_K * (TILE_N / 4); i += 128) {
        int k  = i >> 4;
        int nc = (i & 15) << 2;
        int gk = k0 + k, gn = tileN + nc;
        float4 v = make_float4(0.f, 0.f, 0.f, 0.f);
        if (gk < K) {
          const float* src = &Bm[Boff + (long)gk * ldb + gn];
          if (fullN || gn + 3 < N) {
            v = *(const float4*)src;               // 16B aligned (gn % 4 == 0)
          } else if (gn < N) {                     // tiny-N tail (e.g. N=2)
            v.x = src[0];
            if (gn + 1 < N) v.y = src[1];
            if (gn + 2 < N) v.z = src[2];
          }
        }
        Bs[(nc + 0) * LDSR + k] = (_Float16)v.x;
        Bs[(nc + 1) * LDSR + k] = (_Float16)v.y;
        Bs[(nc + 2) * LDSR + k] = (_Float16)v.z;
        Bs[(nc + 3) * LDSR + k] = (_Float16)v.w;
      }
    }

    // prefetch next K tile (speculative, global_prefetch_b8)
    if (k0 + TILE_K < K) {
      if (tid < 64) {
        int pm = tileM + tid;
        if (pm < M) __builtin_prefetch(&A[Aoff + (long)pm * lda + k0 + TILE_K], 0, 1);
      } else {
        int pn = tileN + (tid - 64);
        if (pn < N) {
          long off = TRANSB ? (Boff + (long)pn * ldb + k0 + TILE_K)
                            : (Boff + (long)(k0 + TILE_K) * ldb + pn);
          __builtin_prefetch(&Bm[off], 0, 1);
        }
      }
    }
    __syncthreads();

    // ---- build fragments per the 16-bit A/B VGPR layout ----
    v16h af[2], bf[2];
#pragma unroll
    for (int i = 0; i < 2; i++) {
      const _Float16* rowA = &As[(wr * 32 + i * 16 + r) * LDSR];
      v8h alo = *(const v8h*)(rowA + 8 * g);
      v8h ahi = *(const v8h*)(rowA + 16 + 8 * g);
#pragma unroll
      for (int e = 0; e < 8; e++) { af[i][e] = alo[e]; af[i][e + 8] = ahi[e]; }
      const _Float16* rowB = &Bs[(wc * 32 + i * 16 + r) * LDSR];
      v8h blo = *(const v8h*)(rowB + 8 * g);
      v8h bhi = *(const v8h*)(rowB + 16 + 8 * g);
#pragma unroll
      for (int e = 0; e < 8; e++) { bf[i][e] = blo[e]; bf[i][e + 8] = bhi[e]; }
    }
#pragma unroll
    for (int i = 0; i < 2; i++)
#pragma unroll
      for (int j = 0; j < 2; j++)
        acc[i][j] = __builtin_amdgcn_wmma_f32_16x16x32_f16(
            false, af[i], false, bf[j], (short)0, acc[i][j], false, false);
    __syncthreads();
  }

  // ---- epilogue (C/D layout: VGPR e -> M=e+8g, N=lane&15) ----
#pragma unroll
  for (int i = 0; i < 2; i++) {
#pragma unroll
    for (int j = 0; j < 2; j++) {
#pragma unroll
      for (int e = 0; e < 8; e++) {
        int row = tileM + wr * 32 + i * 16 + e + 8 * g;
        int col = tileN + wc * 32 + j * 16 + r;
        if (row < M && col < N) {
          long cidx = Coff + (long)row * ldc + col;
          float v = acc[i][j][e] * alpha;
          if (bias)  v += bias[col];
          if (ACCC)  v += C[cidx];
          if (resid) v += resid[cidx];
          if (ACT == 1) v = gelu_f(v);
          if (mask && mask[(long)bb * maskStride + col] == 0) v = -1e9f;
          C[cidx] = v;
        }
      }
    }
  }
}

// ---------------- feature packing ----------------
__global__ __launch_bounds__(256)
void pack_features(const float* __restrict__ xc, const float* __restrict__ sc,
                   const float* __restrict__ tc, const float* __restrict__ fc,
                   const float* __restrict__ xt, const float* __restrict__ st,
                   const float* __restrict__ tt, const float* __restrict__ obs,
                   float* __restrict__ F)
{
  int rr = blockIdx.x * 256 + threadIdx.x;
  if (rr >= NALL) return;
  float* o = F + (long)rr * 16;
  if (rr < NROW) {            // test rows (unobserved -> obs_emb[0])
    int i = rr;
    o[0] = obs[0]; o[1] = obs[1]; o[2] = obs[2]; o[3] = obs[3];
#pragma unroll
    for (int d = 0; d < 8; d++) o[4 + d] = xt[(long)i * 8 + d];
    o[12] = st[(long)i * 2]; o[13] = st[(long)i * 2 + 1];
    o[14] = tt[i];  o[15] = 0.0f;
  } else {                    // ctx rows (observed -> obs_emb[1])
    int i = rr - NROW;
    o[0] = obs[4]; o[1] = obs[5]; o[2] = obs[6]; o[3] = obs[7];
#pragma unroll
    for (int d = 0; d < 8; d++) o[4 + d] = xc[(long)i * 8 + d];
    o[12] = sc[(long)i * 2]; o[13] = sc[(long)i * 2 + 1];
    o[14] = tc[i];  o[15] = fc[i];
  }
}

// ---------------- row LayerNorm over D=128 (one wave32 per row) ----------------
__global__ __launch_bounds__(128)
void layernorm_rows(const float* __restrict__ X, const float* __restrict__ gam,
                    const float* __restrict__ bet, float* __restrict__ Y, int rows)
{
  int row  = blockIdx.x * 4 + (threadIdx.x >> 5);
  int lane = threadIdx.x & 31;
  if (row >= rows) return;
  float4 x4 = *(const float4*)(X + (long)row * Dd + lane * 4);
  float s = x4.x + x4.y + x4.z + x4.w;
#pragma unroll
  for (int o = 16; o > 0; o >>= 1) s += __shfl_xor(s, o, 32);
  float mean = s * (1.0f / 128.0f);
  float dx = x4.x - mean, dy = x4.y - mean, dz = x4.z - mean, dw = x4.w - mean;
  float v = dx * dx + dy * dy + dz * dz + dw * dw;
#pragma unroll
  for (int o = 16; o > 0; o >>= 1) v += __shfl_xor(v, o, 32);
  float rstd = rsqrtf(v * (1.0f / 128.0f) + 1e-5f);
  float4 g4 = *(const float4*)(gam + lane * 4);
  float4 b4 = *(const float4*)(bet + lane * 4);
  float4 y4;
  y4.x = dx * rstd * g4.x + b4.x;  y4.y = dy * rstd * g4.y + b4.y;
  y4.z = dz * rstd * g4.z + b4.z;  y4.w = dw * rstd * g4.w + b4.w;
  *(float4*)(Y + (long)row * Dd + lane * 4) = y4;
}

// ---------------- row softmax over 512 (one wave32 per row) ----------------
__global__ __launch_bounds__(128)
void softmax_rows(float* __restrict__ S, int rows)
{
  int row  = blockIdx.x * 4 + (threadIdx.x >> 5);
  int lane = threadIdx.x & 31;
  if (row >= rows) return;
  float* p = S + (long)row * NKn;
  float vals[16];
  float mx = -3.4e38f;
#pragma unroll
  for (int i = 0; i < 16; i++) { vals[i] = p[lane + i * 32]; mx = fmaxf(mx, vals[i]); }
#pragma unroll
  for (int o = 16; o > 0; o >>= 1) mx = fmaxf(mx, __shfl_xor(mx, o, 32));
  float sum = 0.0f;
#pragma unroll
  for (int i = 0; i < 16; i++) { vals[i] = expf(vals[i] - mx); sum += vals[i]; }
#pragma unroll
  for (int o = 16; o > 0; o >>= 1) sum += __shfl_xor(sum, o, 32);
  float inv = 1.0f / sum;
#pragma unroll
  for (int i = 0; i < 16; i++) p[lane + i * 32] = vals[i] * inv;
}

// ---------------- pairwise -(|q|^2 - 2 q.k + |k|^2) ----------------
template<int DIMS>
__global__ __launch_bounds__(256)
void pairwise(const float* __restrict__ Qv, const float* __restrict__ Kv,
              float* __restrict__ out, int nq, int nk)
{
  int b = blockIdx.z;
  int idx = blockIdx.x * 256 + threadIdx.x;
  if (idx >= nq * nk) return;
  int i = idx / nk, j = idx - i * nk;
  const float* qp = Qv + ((long)b * nq + i) * DIMS;
  const float* kp = Kv + ((long)b * nk + j) * DIMS;
  float qq = 0, kk = 0, qk = 0;
#pragma unroll
  for (int d = 0; d < DIMS; d++) { float a = qp[d], c = kp[d]; qq += a * a; kk += c * c; qk += a * c; }
  out[(long)b * nq * nk + idx] = -(qq - 2.0f * qk + kk);
}

// ---------------- bias MLP: 3 sims -> 8 head biases, writes S[b,h,q,k] ----------------
__global__ __launch_bounds__(256)
void bias_mlp(const float* __restrict__ s0, const float* __restrict__ s1, const float* __restrict__ s2,
              const float* __restrict__ w1a, const float* __restrict__ b1a,
              const float* __restrict__ w2a, const float* __restrict__ b2a,
              const float* __restrict__ w1b, const float* __restrict__ b1b,
              const float* __restrict__ w2b, const float* __restrict__ b2b,
              const float* __restrict__ w1c, const float* __restrict__ b1c,
              const float* __restrict__ w2c, const float* __restrict__ b2c,
              float* __restrict__ S, int nq, int nk)
{
  __shared__ float sw1[3][16], sb1[3][16], sw2[3][128], sb2[3][8];
  int tid = threadIdx.x;
  if (tid < 16) {
    sw1[0][tid] = w1a[tid]; sw1[1][tid] = w1b[tid]; sw1[2][tid] = w1c[tid];
    sb1[0][tid] = b1a[tid]; sb1[1][tid] = b1b[tid]; sb1[2][tid] = b1c[tid];
  }
  if (tid < 128) { sw2[0][tid] = w2a[tid]; sw2[1][tid] = w2b[tid]; sw2[2][tid] = w2c[tid]; }
  if (tid < 8)   { sb2[0][tid] = b2a[tid]; sb2[1][tid] = b2b[tid]; sb2[2][tid] = b2c[tid]; }
  __syncthreads();
  long idx = (long)blockIdx.x * 256 + tid;
  long nqnk = (long)nq * nk;
  if (idx >= (long)Bb * nqnk) return;
  int b    = (int)(idx / nqnk);
  long rem = idx - (long)b * nqnk;
  int qi = (int)(rem / nk);
  int kj = (int)(rem - (long)qi * nk);
  float acc[8];
#pragma unroll
  for (int h = 0; h < 8; h++) acc[h] = 0.0f;
  const float* sims[3] = { s0, s1, s2 };
#pragma unroll
  for (int p = 0; p < 3; p++) {
    float s = sims[p][idx];
    if (s * 0.0f == 0.0f) {              // isfinite
#pragma unroll
      for (int j = 0; j < 16; j++) {
        float hj = gelu_f(fmaf(s, sw1[p][j], sb1[p][j]));
#pragma unroll
        for (int h = 0; h < 8; h++) acc[h] = fmaf(hj, sw2[p][j * 8 + h], acc[h]);
      }
#pragma unroll
      for (int h = 0; h < 8; h++) acc[h] += sb2[p][h];
    }
  }
#pragma unroll
  for (int h = 0; h < 8; h++)
    S[(((long)b * Hh + h) * nq + qi) * (long)nk + kj] = acc[h];
}

// ---------------- head finalize: [loc, softplus(scale)] ----------------
__global__ __launch_bounds__(256)
void finalize_head(const float* __restrict__ G, float* __restrict__ out, int rows)
{
  int r = blockIdx.x * 256 + threadIdx.x;
  if (r >= rows) return;
  float loc = G[2 * r], sc = G[2 * r + 1];
  out[2 * r]     = loc;
  out[2 * r + 1] = log1pf(expf(-fabsf(sc))) + fmaxf(sc, 0.0f);
}

// ---------------- host dispatch ----------------
enum {
  IN_XCTX = 0, IN_SCTX, IN_TCTX, IN_FCTX, IN_XTEST, IN_STEST, IN_TTEST, IN_MASK,
  P_OBS, P_EAW0, P_EAB0, P_EAW1, P_EAB1, P_EAW2, P_EAB2, P_LN0G, P_LN0B,
  KK_LNG, KK_LNB, KK_WQ, KK_WK, KK_WV, KK_WO, KK_FFLNG, KK_FFLNB, KK_FFW1, KK_FFB1, KK_FFW2, KK_FFB2,
  QK_LNG, QK_LNB, QK_WQ, QK_WK, QK_WV, QK_WO, QK_FFLNG, QK_FFLNB, QK_FFW1, QK_FFB1, QK_FFW2, QK_FFB2,
  XB_W1, XB_B1, XB_W2, XB_B2, SB_W1, SB_B1, SB_W2, SB_B2, TB_W1, TB_B1, TB_W2, TB_B2,
  P_LNFG, P_LNFB, P_HW0, P_HB0, P_HW1, P_HB1, P_HW2, P_HB2, N_PARAMS
};

static inline void run_gemm(hipStream_t st, int act, bool scoreMode,
    const float* A, const float* Bm, const float* bias, const float* resid,
    const unsigned char* mask, int maskStride, float* C,
    int M, int N, int K, int lda, int ldb, int ldc,
    long sAb, long sAh, long sBb, long sBh, long sCb, long sCh,
    int nB, int nH, float alpha)
{
  dim3 grid((N + TILE_N - 1) / TILE_N, (M + TILE_M - 1) / TILE_M, nB * nH);
  if (scoreMode)
    gemm_wmma<0, true, true><<<grid, 128, 0, st>>>(A, Bm, bias, resid, mask, maskStride, C,
        M, N, K, lda, ldb, ldc, sAb, sAh, sBb, sBh, sCb, sCh, nH, alpha);
  else if (act == 1)
    gemm_wmma<1, false, false><<<grid, 128, 0, st>>>(A, Bm, bias, resid, mask, maskStride, C,
        M, N, K, lda, ldb, ldc, sAb, sAh, sBb, sBh, sCb, sCh, nH, alpha);
  else
    gemm_wmma<0, false, false><<<grid, 128, 0, st>>>(A, Bm, bias, resid, mask, maskStride, C,
        M, N, K, lda, ldb, ldc, sAb, sAh, sBb, sBh, sCb, sCh, nH, alpha);
}

extern "C" void kernel_launch(void* const* d_in, const int* in_sizes, int n_in,
                              void* d_out, int out_size, void* d_ws, size_t ws_size,
                              hipStream_t stream)
{
  (void)in_sizes; (void)out_size; (void)ws_size;
  const float* P[N_PARAMS];
  for (int i = 0; i < n_in && i < N_PARAMS; i++) P[i] = (const float*)d_in[i];
  const unsigned char* mask = (const unsigned char*)d_in[IN_MASK];

  // workspace carve-up (floats); total ~19M floats (~76MB)
  float* W  = (float*)d_ws;
  float* F  = W;                               // [4096,16]
  float* X  = F  + (size_t)NALL * 16;          // [4096,128]  rows 0..2047 = qvs, 2048.. = kvs
  float* LN = X  + (size_t)NALL * Dd;          // [4096,128]
  float* H1 = LN + (size_t)NALL * Dd;          // [4096,256] / FF hidden [2048,512]
  float* H2 = H1 + (size_t)NALL * 256;         // [4096,256]
  float* Qb = H2 + (size_t)NALL * 256;         // [2048,128]
  float* Kb = Qb + (size_t)NROW * Dd;
  float* Vb = Kb + (size_t)NROW * Dd;
  float* Ob = Vb + (size_t)NROW * Dd;
  float* G  = Ob + (size_t)NROW * Dd;          // [2048,2]
  float* S  = G  + (size_t)NROW * 2;           // [4,8,512,512]
  float* SIM = S + (size_t)Bb * Hh * NQn * NKn; // 6 x [4,512,512]: xq,sq,tq,xk,sk,tk
  const size_t SIMSZ = (size_t)Bb * NQn * NKn;
  float* kvs = X + (size_t)NROW * Dd;

  // ---- embed both populations at once ----
  pack_features<<<NALL / 256, 256, 0, stream>>>(P[IN_XCTX], P[IN_SCTX], P[IN_TCTX], P[IN_FCTX],
                                                P[IN_XTEST], P[IN_STEST], P[IN_TTEST], P[P_OBS], F);
  run_gemm(stream, 1, false, F,  P[P_EAW0], P[P_EAB0], nullptr, nullptr, 0, H1,
           NALL, 256, 16, 16, 256, 256, 0,0,0,0,0,0, 1, 1, 1.0f);
  run_gemm(stream, 1, false, H1, P[P_EAW1], P[P_EAB1], nullptr, nullptr, 0, H2,
           NALL, 256, 256, 256, 256, 256, 0,0,0,0,0,0, 1, 1, 1.0f);
  run_gemm(stream, 0, false, H2, P[P_EAW2], P[P_EAB2], nullptr, nullptr, 0, X,
           NALL, Dd, 256, 256, Dd, Dd, 0,0,0,0,0,0, 1, 1, 1.0f);
  layernorm_rows<<<NALL / 4, 128, 0, stream>>>(X, P[P_LN0G], P[P_LN0B], X, NALL);

  // ---- pairwise similarity maps ----
  {
    dim3 pg((NQn * NKn) / 256, 1, Bb);
    pairwise<8><<<pg, 256, 0, stream>>>(P[IN_XTEST], P[IN_XCTX], SIM + 0 * SIMSZ, NQn, NKn);
    pairwise<2><<<pg, 256, 0, stream>>>(P[IN_STEST], P[IN_SCTX], SIM + 1 * SIMSZ, NQn, NKn);
    pairwise<1><<<pg, 256, 0, stream>>>(P[IN_TTEST], P[IN_TCTX], SIM + 2 * SIMSZ, NQn, NKn);
    pairwise<8><<<pg, 256, 0, stream>>>(P[IN_XCTX],  P[IN_XCTX], SIM + 3 * SIMSZ, NKn, NKn);
    pairwise<2><<<pg, 256, 0, stream>>>(P[IN_SCTX],  P[IN_SCTX], SIM + 4 * SIMSZ, NKn, NKn);
    pairwise<1><<<pg, 256, 0, stream>>>(P[IN_TCTX],  P[IN_TCTX], SIM + 5 * SIMSZ, NKn, NKn);
  }

  // ---- transformer layers ----
  for (int l = 0; l < Lb; l++) {
    for (int side = 0; side < 2; side++) {
      bool isKK = (side == 0);
      const float* lng   = P[isKK ? KK_LNG : QK_LNG] + (size_t)l * Dd;
      const float* lnb   = P[isKK ? KK_LNB : QK_LNB] + (size_t)l * Dd;
      const float* wq    = P[isKK ? KK_WQ  : QK_WQ]  + (size_t)l * Dd * Dd;
      const float* wk    = P[isKK ? KK_WK  : QK_WK]  + (size_t)l * Dd * Dd;
      const float* wv    = P[isKK ? KK_WV  : QK_WV]  + (size_t)l * Dd * Dd;
      const float* wo    = P[isKK ? KK_WO  : QK_WO]  + (size_t)l * Dd * Dd;
      const float* fflng = P[isKK ? KK_FFLNG : QK_FFLNG] + (size_t)l * Dd;
      const float* fflnb = P[isKK ? KK_FFLNB : QK_FFLNB] + (size_t)l * Dd;
      const float* ffw1  = P[isKK ? KK_FFW1 : QK_FFW1] + (size_t)l * Dd * DFFd;
      const float* ffb1  = P[isKK ? KK_FFB1 : QK_FFB1] + (size_t)l * DFFd;
      const float* ffw2  = P[isKK ? KK_FFW2 : QK_FFW2] + (size_t)l * DFFd * Dd;
      const float* ffb2  = P[isKK ? KK_FFB2 : QK_FFB2] + (size_t)l * Dd;
      float* xbase = isKK ? kvs : X;            // residual stream being updated
      int simBase  = isKK ? 3 : 0;

      // per-head additive bias -> S[b,h,q,k]
      bias_mlp<<<(int)((Bb * (long)NQn * NKn) / 256), 256, 0, stream>>>(
          SIM + (simBase + 0) * SIMSZ, SIM + (simBase + 1) * SIMSZ, SIM + (simBase + 2) * SIMSZ,
          P[XB_W1] + (size_t)l * BHh, P[XB_B1] + (size_t)l * BHh,
          P[XB_W2] + (size_t)l * BHh * Hh, P[XB_B2] + (size_t)l * Hh,
          P[SB_W1] + (size_t)l * BHh, P[SB_B1] + (size_t)l * BHh,
          P[SB_W2] + (size_t)l * BHh * Hh, P[SB_B2] + (size_t)l * Hh,
          P[TB_W1] + (size_t)l * BHh, P[TB_B1] + (size_t)l * BHh,
          P[TB_W2] + (size_t)l * BHh * Hh, P[TB_B2] + (size_t)l * Hh,
          S, NQn, NKn);

      // pre-LN; KK uses LN(h) for both q and kv; QK uses LN(qvs) vs raw kvs
      layernorm_rows<<<NROW / 4, 128, 0, stream>>>(xbase, lng, lnb, LN, NROW);
      const float* qin  = LN;
      const float* kvin = isKK ? LN : kvs;

      run_gemm(stream, 0, false, qin,  wq, nullptr, nullptr, nullptr, 0, Qb,
               NROW, Dd, Dd, Dd, Dd, Dd, 0,0,0,0,0,0, 1, 1, 1.0f);
      run_gemm(stream, 0, false, kvin, wk, nullptr, nullptr, nullptr, 0, Kb,
               NROW, Dd, Dd, Dd, Dd, Dd, 0,0,0,0,0,0, 1, 1, 1.0f);
      run_gemm(stream, 0, false, kvin, wv, nullptr, nullptr, nullptr, 0, Vb,
               NROW, Dd, Dd, Dd, Dd, Dd, 0,0,0,0,0,0, 1, 1, 1.0f);

      // S = bias + scale * Q @ K^T, key-masked; batched over (b,h)
      run_gemm(stream, 0, true, Qb, Kb, nullptr, nullptr, mask, NKn, S,
               NQn, NKn, DHd, Dd, Dd, NKn,
               (long)NQn * Dd, DHd, (long)NKn * Dd, DHd,
               (long)Hh * NQn * NKn, (long)NQn * NKn, Bb, Hh, 0.25f);
      softmax_rows<<<(Bb * Hh * NQn) / 4, 128, 0, stream>>>(S, Bb * Hh * NQn);
      // O = A @ V
      run_gemm(stream, 0, false, S, Vb, nullptr, nullptr, nullptr, 0, Ob,
               NQn, DHd, NKn, NKn, Dd, Dd,
               (long)Hh * NQn * NKn, (long)NQn * NKn,
               (long)NKn * Dd, DHd, (long)NQn * Dd, DHd, Bb, Hh, 1.0f);
      // x += O @ wo
      run_gemm(stream, 0, false, Ob, wo, nullptr, xbase, nullptr, 0, xbase,
               NROW, Dd, Dd, Dd, Dd, Dd, 0,0,0,0,0,0, 1, 1, 1.0f);

      // FFN: x += gelu(LN(x) @ w1 + b1) @ w2 + b2
      layernorm_rows<<<NROW / 4, 128, 0, stream>>>(xbase, fflng, fflnb, LN, NROW);
      run_gemm(stream, 1, false, LN, ffw1, ffb1, nullptr, nullptr, 0, H1,
               NROW, DFFd, Dd, Dd, DFFd, DFFd, 0,0,0,0,0,0, 1, 1, 1.0f);
      run_gemm(stream, 0, false, H1, ffw2, ffb2, xbase, nullptr, 0, xbase,
               NROW, Dd, DFFd, DFFd, Dd, Dd, 0,0,0,0,0,0, 1, 1, 1.0f);
    }
  }

  // ---- output head ----
  layernorm_rows<<<NROW / 4, 128, 0, stream>>>(X, P[P_LNFG], P[P_LNFB], LN, NROW);
  run_gemm(stream, 1, false, LN, P[P_HW0], P[P_HB0], nullptr, nullptr, 0, H1,
           NROW, 256, Dd, Dd, 256, 256, 0,0,0,0,0,0, 1, 1, 1.0f);
  run_gemm(stream, 1, false, H1, P[P_HW1], P[P_HB1], nullptr, nullptr, 0, H2,
           NROW, 64, 256, 256, 64, 64, 0,0,0,0,0,0, 1, 1, 1.0f);
  run_gemm(stream, 0, false, H2, P[P_HW2], P[P_HB2], nullptr, nullptr, 0, G,
           NROW, 2, 64, 64, 2, 2, 0,0,0,0,0,0, 1, 1, 1.0f);
  finalize_head<<<NROW / 256, 256, 0, stream>>>(G, (float*)d_out, NROW);
}
